// MyPointAggregateBlock_32581621907888
// MI455X (gfx1250) — matrix operand order repaired and checked by reading, not compiled, not run on hardware
//
#include <hip/hip_runtime.h>
#include <hip/hip_bf16.h>

typedef _Float16 v16h __attribute__((ext_vector_type(16)));
typedef _Float16 v8h  __attribute__((ext_vector_type(8)));
typedef _Float16 v4h  __attribute__((ext_vector_type(4)));
typedef float    v8f  __attribute__((ext_vector_type(8)));

#define H       128
#define NROWS   300000
#define NLANES  25000
#define TM      64          // rows per block
#define BLOCKT  128         // 4 waves of 32

union AFrag { v16h v; v8h h[2]; };

// ---------------- fragment loaders ----------------

// A fragment (16x32 f16) from row-major f16 LDS tile, row stride = strideH halves.
// ISA layout: lanes 0-15 (M=lane): VGPR0-3 K=0..7, VGPR4-7 K=16..23
//             lanes 16-31 (M=lane-16): VGPR0-3 K=8..15, VGPR4-7 K=24..31
__device__ __forceinline__ v16h load_afrag(const _Float16* sA, int strideH,
                                           int rowbase, int lane, int kt) {
    int m  = rowbase + (lane & 15);
    int k0 = kt * 32 + ((lane >> 4) ? 8 : 0);
    const _Float16* p = sA + m * strideH + k0;
    AFrag f;
    f.h[0] = *(const v8h*)(p);        // K k0..k0+7
    f.h[1] = *(const v8h*)(p + 16);   // K k0+16..k0+23
    return f.v;
}

// B fragment: preconverted fragment-ordered buffer, 32 lanes x 16 halves contiguous.
__device__ __forceinline__ v16h load_bfrag(const _Float16* wf, int kt, int nt, int lane) {
    return *(const v16h*)(wf + ((((kt << 3) + nt) * 32 + lane) << 4));
}

// One wave computes a 16xH output tile: sC[rowbase..rowbase+15][0..127] = A @ W + bias
template<int KT>
__device__ __forceinline__ void wave_gemm(const _Float16* sA, int strideH, int rowbase,
                                          const _Float16* wf, const float* __restrict__ bias,
                                          float* sC, int lane) {
    v16h a[KT];
#pragma unroll
    for (int kt = 0; kt < KT; ++kt) a[kt] = load_afrag(sA, strideH, rowbase, lane, kt);
    int ncol_lo = lane & 15;
    int rbase   = rowbase + ((lane >> 4) << 3);
#pragma unroll
    for (int nt = 0; nt < 8; ++nt) {
        float bv = bias[nt * 16 + ncol_lo];
        v8f c = {bv, bv, bv, bv, bv, bv, bv, bv};
#pragma unroll
        for (int kt = 0; kt < KT; ++kt) {
            v16h b = load_bfrag(wf, kt, nt, lane);
            c = __builtin_amdgcn_wmma_f32_16x16x32_f16(
                    false, a[kt], false, b, (short)0, c, false, false);
        }
#pragma unroll
        for (int r = 0; r < 8; ++r)
            sC[(rbase + r) * H + nt * 16 + ncol_lo] = c[r];
    }
}

// Row-wise LayerNorm + ReLU over 128 f32 values, write f16 row (stride in caller).
__device__ __forceinline__ void ln_relu_to_f16(const float* sCrow, _Float16* sArow,
                                               const float* __restrict__ g,
                                               const float* __restrict__ be) {
    float s = 0.f, s2 = 0.f;
#pragma unroll
    for (int c = 0; c < H; c += 4) {
        float4 x = *(const float4*)(sCrow + c);
        s  += x.x + x.y + x.z + x.w;
        s2 += x.x * x.x + x.y * x.y + x.z * x.z + x.w * x.w;
    }
    float m   = s * (1.f / H);
    float var = s2 * (1.f / H) - m * m;
    float rs  = rsqrtf(var + 1e-5f);
#pragma unroll
    for (int c = 0; c < H; c += 4) {
        float4 x  = *(const float4*)(sCrow + c);
        float4 gv = *(const float4*)(g + c);
        float4 bv = *(const float4*)(be + c);
        float y0 = fmaxf((x.x - m) * rs * gv.x + bv.x, 0.f);
        float y1 = fmaxf((x.y - m) * rs * gv.y + bv.y, 0.f);
        float y2 = fmaxf((x.z - m) * rs * gv.z + bv.z, 0.f);
        float y3 = fmaxf((x.w - m) * rs * gv.w + bv.w, 0.f);
        v4h hv = {(_Float16)y0, (_Float16)y1, (_Float16)y2, (_Float16)y3};
        *(v4h*)(sArow + c) = hv;
    }
}

// ---------------- kernels ----------------

__global__ void k_zero(unsigned* p, int n) {
    int i = blockIdx.x * 256 + threadIdx.x;
    if (i < n) p[i] = 0u;
}

// Convert fp32 row-major [K][128] weights to B-fragment-ordered f16.
// grid = (8 ntiles, ktiles), 32 threads.
__global__ void k_wconv(const float* __restrict__ w, _Float16* __restrict__ wf) {
    int lane = threadIdx.x;
    int nt = blockIdx.x, kt = blockIdx.y;
    int n = nt * 16 + (lane & 15);
    int kbase = kt * 32 + ((lane >> 4) ? 8 : 0);
    _Float16* dst = wf + ((((kt << 3) + nt) * 32 + lane) << 4);
#pragma unroll
    for (int j = 0; j < 8; ++j) {
        dst[j]     = (_Float16)w[(kbase + j) * H + n];
        dst[j + 8] = (_Float16)w[(kbase + 16 + j) * H + n];
    }
}

__global__ __launch_bounds__(BLOCKT) void k_fc1(
    const float* __restrict__ x, const int* __restrict__ lane_ids,
    const _Float16* __restrict__ w1a_f, const float* __restrict__ b1a,
    const float* __restrict__ g1a, const float* __restrict__ be1a,
    const _Float16* __restrict__ w1b_f, const float* __restrict__ b1b,
    const float* __restrict__ g1b, const float* __restrict__ be1b,
    _Float16* __restrict__ hout, unsigned* __restrict__ maxbuf)
{
    __shared__ __align__(16) _Float16 sA[TM * H];
    __shared__ __align__(16) float    sC[TM * H];
    int tid  = threadIdx.x;
    int lane = tid & 31;
    int wave = tid >> 5;
    size_t row0 = (size_t)blockIdx.x * TM;

    // stage x -> f16 LDS tile (coalesced float4 loads)
    for (int i = tid; i < TM * (H / 4); i += BLOCKT) {
        int r  = i >> 5;
        int c4 = (i & 31) << 2;
        size_t gr = row0 + r;
        float4 xv = (gr < NROWS) ? *(const float4*)(x + gr * H + c4)
                                 : make_float4(0.f, 0.f, 0.f, 0.f);
        v4h hv = {(_Float16)xv.x, (_Float16)xv.y, (_Float16)xv.z, (_Float16)xv.w};
        *(v4h*)(sA + r * H + c4) = hv;
    }
    __syncthreads();

    int rowbase = wave * 16;
    wave_gemm<4>(sA, H, rowbase, w1a_f, b1a, sC, lane);
    __syncthreads();
    if (lane < 16)
        ln_relu_to_f16(sC + (rowbase + lane) * H, sA + (rowbase + lane) * H, g1a, be1a);
    __syncthreads();
    wave_gemm<4>(sA, H, rowbase, w1b_f, b1b, sC, lane);
    __syncthreads();
    if (lane < 16)
        ln_relu_to_f16(sC + (rowbase + lane) * H, sA + (rowbase + lane) * H, g1b, be1b);
    __syncthreads();

    // write h (f16, coalesced) + scatter-max into lane buffer (u32 max == f32 max for >=0)
    for (int i = tid; i < TM * (H / 4); i += BLOCKT) {
        int r  = i >> 5;
        int c4 = (i & 31) << 2;
        size_t gr = row0 + r;
        if (gr < NROWS) {
            *(uint2*)(hout + gr * H + c4) = *(const uint2*)(sA + r * H + c4);
            int lid = lane_ids[gr];
            unsigned* mb = maxbuf + (size_t)lid * H + c4;
#pragma unroll
            for (int j = 0; j < 4; ++j) {
                float v = (float)sA[r * H + c4 + j];
                atomicMax(mb + j, __float_as_uint(v));
            }
        }
    }
}

__global__ __launch_bounds__(BLOCKT) void k_fc2(
    const float* __restrict__ x, const int* __restrict__ lane_ids,
    const _Float16* __restrict__ h, const unsigned* __restrict__ maxbuf,
    const _Float16* __restrict__ w2a_f, const float* __restrict__ b2a,
    const float* __restrict__ g2a, const float* __restrict__ be2a,
    const _Float16* __restrict__ w2b_f, const float* __restrict__ b2b,
    const float* __restrict__ g2b, const float* __restrict__ be2b,
    const float* __restrict__ gn, const float* __restrict__ bn,
    float* __restrict__ out)
{
    __shared__ __align__(16) _Float16 sA[TM * 2 * H];   // [64][256] f16
    __shared__ __align__(16) float    sC[TM * H];       // [64][128] f32
    int tid  = threadIdx.x;
    int lane = tid & 31;
    int wave = tid >> 5;
    size_t row0 = (size_t)blockIdx.x * TM;

    // stage A = [h | gather(maxbuf)] as f16 [64][256]
    for (int i = tid; i < TM * (H / 4); i += BLOCKT) {
        int r  = i >> 5;
        int c4 = (i & 31) << 2;
        size_t gr = row0 + r;
        if (gr < NROWS) {
            *(uint2*)(sA + r * 2 * H + c4) = *(const uint2*)(h + gr * H + c4);
            int lid = lane_ids[gr];
            uint4 mv = *(const uint4*)(maxbuf + (size_t)lid * H + c4);
            v4h hv = {(_Float16)__uint_as_float(mv.x), (_Float16)__uint_as_float(mv.y),
                      (_Float16)__uint_as_float(mv.z), (_Float16)__uint_as_float(mv.w)};
            *(v4h*)(sA + r * 2 * H + H + c4) = hv;
        } else {
            uint2 z = {0u, 0u};
            *(uint2*)(sA + r * 2 * H + c4) = z;
            *(uint2*)(sA + r * 2 * H + H + c4) = z;
        }
    }
    __syncthreads();

    int rowbase = wave * 16;
    wave_gemm<8>(sA, 2 * H, rowbase, w2a_f, b2a, sC, lane);   // K = 256
    __syncthreads();
    if (lane < 16)
        ln_relu_to_f16(sC + (rowbase + lane) * H, sA + (rowbase + lane) * 2 * H, g2a, be2a);
    __syncthreads();
    wave_gemm<4>(sA, 2 * H, rowbase, w2b_f, b2b, sC, lane);   // K = 128 (cols 0..127)
    __syncthreads();

    // stage x into f32 alias of sA (A no longer needed)
    float* sX = (float*)sA;   // [64][128] f32, same 32KB
    for (int i = tid; i < TM * (H / 4); i += BLOCKT) {
        int r  = i >> 5;
        int c4 = (i & 31) << 2;
        size_t gr = row0 + r;
        *(float4*)(sX + r * H + c4) = (gr < NROWS) ? *(const float4*)(x + gr * H + c4)
                                                   : make_float4(0.f, 0.f, 0.f, 0.f);
    }
    __syncthreads();

    if (lane < 16) {
        const float* t  = sC + (rowbase + lane) * H;   // GEMM2 output row
        float*       sx = sX + (rowbase + lane) * H;   // x row -> s row -> out row
        // LN2b stats
        float s = 0.f, s2 = 0.f;
#pragma unroll
        for (int c = 0; c < H; c += 4) {
            float4 v = *(const float4*)(t + c);
            s  += v.x + v.y + v.z + v.w;
            s2 += v.x * v.x + v.y * v.y + v.z * v.z + v.w * v.w;
        }
        float m  = s * (1.f / H);
        float rs = rsqrtf(s2 * (1.f / H) - m * m + 1e-5f);
        // f = relu(ln(t)); sxc = x + f; accumulate final stats
        float u = 0.f, u2 = 0.f;
#pragma unroll
        for (int c = 0; c < H; ++c) {
            float f = fmaxf((t[c] - m) * rs * g2b[c] + be2b[c], 0.f);
            float sv = sx[c] + f;
            sx[c] = sv;
            u  += sv;
            u2 += sv * sv;
        }
        float m2  = u * (1.f / H);
        float rs2 = rsqrtf(u2 * (1.f / H) - m2 * m2 + 1e-5f);
#pragma unroll
        for (int c = 0; c < H; ++c)
            sx[c] = (sx[c] - m2) * rs2 * gn[c] + bn[c];
    }
    __syncthreads();

    for (int i = tid; i < TM * (H / 4); i += BLOCKT) {
        int r  = i >> 5;
        int c4 = (i & 31) << 2;
        size_t gr = row0 + r;
        if (gr < NROWS)
            *(float4*)(out + gr * H + c4) = *(const float4*)(sX + r * H + c4);
    }
}

// ---------------- launch ----------------

extern "C" void kernel_launch(void* const* d_in, const int* in_sizes, int n_in,
                              void* d_out, int out_size, void* d_ws, size_t ws_size,
                              hipStream_t stream) {
    const float* x        = (const float*)d_in[0];
    const int*   lane_ids = (const int*)d_in[1];
    const float* w1a = (const float*)d_in[2];
    const float* b1a = (const float*)d_in[3];
    const float* g1a = (const float*)d_in[4];
    const float* be1a = (const float*)d_in[5];
    const float* w1b = (const float*)d_in[6];
    const float* b1b = (const float*)d_in[7];
    const float* g1b = (const float*)d_in[8];
    const float* be1b = (const float*)d_in[9];
    const float* w2a = (const float*)d_in[10];
    const float* b2a = (const float*)d_in[11];
    const float* g2a = (const float*)d_in[12];
    const float* be2a = (const float*)d_in[13];
    const float* w2b = (const float*)d_in[14];
    const float* b2b = (const float*)d_in[15];
    const float* g2b = (const float*)d_in[16];
    const float* be2b = (const float*)d_in[17];
    const float* gn = (const float*)d_in[18];
    const float* bn = (const float*)d_in[19];
    float* out = (float*)d_out;

    char* ws = (char*)d_ws;
    size_t off = 0;
    _Float16* h16 = (_Float16*)(ws + off);      off += (size_t)NROWS * H * sizeof(_Float16);
    unsigned* maxbuf = (unsigned*)(ws + off);   off += (size_t)NLANES * H * sizeof(unsigned);
    _Float16* w1a_f = (_Float16*)(ws + off);    off += (size_t)H * H * sizeof(_Float16);
    _Float16* w1b_f = (_Float16*)(ws + off);    off += (size_t)H * H * sizeof(_Float16);
    _Float16* w2a_f = (_Float16*)(ws + off);    off += (size_t)2 * H * H * sizeof(_Float16);
    _Float16* w2b_f = (_Float16*)(ws + off);    off += (size_t)H * H * sizeof(_Float16);
    if (ws_size < off) return;

    k_zero<<<(NLANES * H + 255) / 256, 256, 0, stream>>>(maxbuf, NLANES * H);
    k_wconv<<<dim3(8, 4), 32, 0, stream>>>(w1a, w1a_f);
    k_wconv<<<dim3(8, 4), 32, 0, stream>>>(w1b, w1b_f);
    k_wconv<<<dim3(8, 8), 32, 0, stream>>>(w2a, w2a_f);
    k_wconv<<<dim3(8, 4), 32, 0, stream>>>(w2b, w2b_f);

    int nblocks = (NROWS + TM - 1) / TM;
    k_fc1<<<nblocks, BLOCKT, 0, stream>>>(x, lane_ids,
                                          w1a_f, b1a, g1a, be1a,
                                          w1b_f, b1b, g1b, be1b,
                                          h16, maxbuf);
    k_fc2<<<nblocks, BLOCKT, 0, stream>>>(x, lane_ids, h16, maxbuf,
                                          w2a_f, b2a, g2a, be2a,
                                          w2b_f, b2b, g2b, be2b,
                                          gn, bn, out);
}
